// BotRGCN_48404281426130
// MI455X (gfx1250) — matrix operand order repaired and checked by reading, not compiled, not run on hardware
//
#include <hip/hip_runtime.h>
#include <hip/hip_bf16.h>
#include <stdint.h>

#define NN 50000       // nodes
#define NE 600000      // edges
#define NR 3           // relations
#define FD 128         // feature dim
#define NI 10000       // output rows
#define KFUSE 800      // 16 (value) + 768 (text) + 16 pad  -> 25 k-steps of 32

typedef __attribute__((ext_vector_type(16))) __bf16 v16bf;
typedef __attribute__((ext_vector_type(8)))  float  v8f;
typedef __attribute__((ext_vector_type(8)))  unsigned int v8u;

// ---------- bf16 helpers ----------
__device__ __forceinline__ unsigned short f2bf(float f) {
  unsigned u = __builtin_bit_cast(unsigned, f);
  u += 0x7FFFu + ((u >> 16) & 1u);          // round-to-nearest-even
  return (unsigned short)(u >> 16);
}
__device__ __forceinline__ unsigned pk2(float lo, float hi) {
  return (unsigned)f2bf(lo) | ((unsigned)f2bf(hi) << 16);
}
__device__ __forceinline__ v16bf load_b_frag(const unsigned* __restrict__ bp) {
  uint4 q0 = ((const uint4*)bp)[0];
  uint4 q1 = ((const uint4*)bp)[1];
  v8u u;
  u[0]=q0.x; u[1]=q0.y; u[2]=q0.z; u[3]=q0.w;
  u[4]=q1.x; u[5]=q1.y; u[6]=q1.z; u[7]=q1.w;
  return __builtin_bit_cast(v16bf, u);
}

// ---------- weight preparation ----------
// Wc[k,e]: k<16 -> fc1_w @ relu_w_top ; 16<=k<784 -> fc2_w @ relu_w_bot ; k>=784 -> 0
__global__ void fuse_weights(const float* __restrict__ fc1_w, const float* __restrict__ fc1_b,
                             const float* __restrict__ fc2_w, const float* __restrict__ fc2_b,
                             const float* __restrict__ relu_w, const float* __restrict__ relu_b,
                             float* __restrict__ Wc, float* __restrict__ biasc) {
  int t = blockIdx.x * blockDim.x + threadIdx.x;
  if (t < KFUSE * FD) {
    int k = t / FD, e = t % FD;
    float s = 0.f;
    if (k < 16) {
      for (int d = 0; d < FD; ++d) s += fc1_w[k * FD + d] * relu_w[d * FD + e];
    } else if (k < 784) {
      int j = k - 16;
      for (int d = 0; d < FD; ++d) s += fc2_w[j * FD + d] * relu_w[(FD + d) * FD + e];
    }
    Wc[t] = s;
  } else if (t < KFUSE * FD + FD) {
    int e = t - KFUSE * FD;
    float s = relu_b[e];
    for (int d = 0; d < FD; ++d)
      s += fc1_b[d] * relu_w[d * FD + e] + fc2_b[d] * relu_w[(FD + d) * FD + e];
    biasc[e] = s;
  }
}

// Repack row-major f32 B[K x 128] into per-(kstep,colblock,lane) 8-dword bf16 fragments.
__global__ void pack_b(const float* __restrict__ B, unsigned* __restrict__ out, int K) {
  int t = blockIdx.x * blockDim.x + threadIdx.x;
  int total = (K / 32) * 2048;                    // (K/32)*8cb*32lane*8dw
  if (t >= total) return;
  int j    = t & 7;
  int lane = (t >> 3) & 31;
  int cb   = (t >> 8) & 7;
  int kk   = t >> 11;
  int k    = kk * 32 + ((j & 4) ? 16 : 0) + ((lane & 16) ? 8 : 0) + 2 * (j & 3);
  int col  = cb * 16 + (lane & 15);
  out[t] = pk2(B[(size_t)k * FD + col], B[(size_t)(k + 1) * FD + col]);
}

// ---------- degree ----------
__global__ void zero_f32(float* __restrict__ p, int n) {
  int i = blockIdx.x * blockDim.x + threadIdx.x;
  if (i < n) p[i] = 0.f;
}
__global__ void deg_count(const int* __restrict__ dst, const int* __restrict__ et,
                          float* __restrict__ deg, int E) {
  int e = blockIdx.x * blockDim.x + threadIdx.x;
  if (e < E) atomicAdd(&deg[(size_t)et[e] * NN + dst[e]], 1.0f);
}
__global__ void deg_inv(float* __restrict__ deg, int n) {
  int i = blockIdx.x * blockDim.x + threadIdx.x;
  if (i < n) deg[i] = 1.0f / fmaxf(deg[i], 1.0f);
}

// ---------- fused encoder GEMM: h = leakyrelu([value|text] @ Wc + biasc) ----------
__device__ __forceinline__ void load8_fused(const float* __restrict__ value,
                                            const float* __restrict__ text,
                                            int row, int kb, unsigned* d) {
  const float* p;
  if (kb < 16)       p = value + (size_t)row * 16 + kb;
  else if (kb < 784) p = text + (size_t)row * 768 + (kb - 16);
  else { d[0] = d[1] = d[2] = d[3] = 0u; return; }
  float4 f0 = *(const float4*)p;
  float4 f1 = *(const float4*)(p + 4);
  d[0] = pk2(f0.x, f0.y); d[1] = pk2(f0.z, f0.w);
  d[2] = pk2(f1.x, f1.y); d[3] = pk2(f1.z, f1.w);
}

__device__ __forceinline__ v16bf load_a_fused(const float* __restrict__ value,
                                              const float* __restrict__ text,
                                              int row, int k0, int lane, bool rv) {
  v8u au;
  if (rv) {
    unsigned d[8];
    load8_fused(value, text, row, k0 + ((lane & 16) ? 8 : 0), d);
    load8_fused(value, text, row, k0 + ((lane & 16) ? 24 : 16), d + 4);
    for (int i = 0; i < 8; ++i) au[i] = d[i];
  } else {
    for (int i = 0; i < 8; ++i) au[i] = 0u;
  }
  return __builtin_bit_cast(v16bf, au);
}

__global__ __launch_bounds__(256) void gemm_fused_first(
    const float* __restrict__ value, const float* __restrict__ text,
    const unsigned* __restrict__ Bp, const float* __restrict__ biasc,
    float* __restrict__ h, unsigned short* __restrict__ hbf, int M) {
  const int lane = threadIdx.x & 31;
  const int wave = threadIdx.x >> 5;
  const int r0 = blockIdx.x * 128 + wave * 16;
  const int rit = lane & 15;
  const int row = r0 + rit;
  const bool rv = row < M;
  v8f acc[8];
  for (int cb = 0; cb < 8; ++cb)
    for (int i = 0; i < 8; ++i) acc[cb][i] = 0.f;

  const int KS = KFUSE / 32;
  v16bf a = load_a_fused(value, text, row, 0, lane, rv);
  for (int kk = 0; kk < KS; ++kk) {
    // batch all 8 B fragments: one load clause, progressive waits before each WMMA
    v16bf b[8];
#pragma unroll
    for (int cb = 0; cb < 8; ++cb)
      b[cb] = load_b_frag(Bp + (((size_t)(kk * 8 + cb)) * 32 + lane) * 8);
    // software-pipeline A for next k-step
    v16bf a_next = (kk + 1 < KS) ? load_a_fused(value, text, row, (kk + 1) * 32, lane, rv) : a;
#pragma unroll
    for (int cb = 0; cb < 8; ++cb)
      acc[cb] = __builtin_amdgcn_wmma_f32_16x16x32_bf16(
          false, a, false, b[cb], (short)0, acc[cb], false, false);
    a = a_next;
  }
  const int moff = (lane & 16) ? 8 : 0;
  for (int cb = 0; cb < 8; ++cb) {
    int col = cb * 16 + rit;
    float bia = biasc[col];
    for (int v = 0; v < 8; ++v) {
      int r = r0 + moff + v;
      if (r < M) {
        float x = acc[cb][v] + bia;
        x = (x > 0.f) ? x : 0.01f * x;            // leaky relu
        h[(size_t)r * FD + col] = x;
        hbf[(size_t)r * FD + col] = f2bf(x);
      }
    }
  }
}

// ---------- RGCN transforms: z<3 -> xw[z] = h @ w_rel[z] ; z==3 -> agg = h @ w_root + b ----------
__device__ __forceinline__ v16bf load_a_bf16(const unsigned short* __restrict__ A,
                                             int row, int k0, int lane, bool rv) {
  v8u au;
  if (rv) {
    int kb0 = k0 + ((lane & 16) ? 8 : 0);
    int kb1 = k0 + ((lane & 16) ? 24 : 16);
    uint4 q0 = *(const uint4*)(A + (size_t)row * FD + kb0);
    uint4 q1 = *(const uint4*)(A + (size_t)row * FD + kb1);
    au[0] = q0.x; au[1] = q0.y; au[2] = q0.z; au[3] = q0.w;
    au[4] = q1.x; au[5] = q1.y; au[6] = q1.z; au[7] = q1.w;
  } else {
    for (int i = 0; i < 8; ++i) au[i] = 0u;
  }
  return __builtin_bit_cast(v16bf, au);
}

__global__ __launch_bounds__(256) void gemm_rgcn(
    const unsigned short* __restrict__ hbf, const unsigned* __restrict__ Bp4,
    const float* __restrict__ bias, float* __restrict__ xw, float* __restrict__ agg, int M) {
  const int z = blockIdx.z;
  const unsigned* Bp = Bp4 + (size_t)z * 8192;    // 4 ksteps * 2048 uints
  float* C = (z < 3) ? (xw + (size_t)z * M * FD) : agg;
  const int lane = threadIdx.x & 31;
  const int wave = threadIdx.x >> 5;
  const int r0 = blockIdx.x * 128 + wave * 16;
  const int rit = lane & 15;
  const int row = r0 + rit;
  const bool rv = row < M;
  v8f acc[8];
  for (int cb = 0; cb < 8; ++cb)
    for (int i = 0; i < 8; ++i) acc[cb][i] = 0.f;

  const int KS = FD / 32;
  v16bf a = load_a_bf16(hbf, row, 0, lane, rv);
#pragma unroll
  for (int kk = 0; kk < KS; ++kk) {
    v16bf b[8];
#pragma unroll
    for (int cb = 0; cb < 8; ++cb)
      b[cb] = load_b_frag(Bp + (((size_t)(kk * 8 + cb)) * 32 + lane) * 8);
    v16bf a_next = (kk + 1 < KS) ? load_a_bf16(hbf, row, (kk + 1) * 32, lane, rv) : a;
#pragma unroll
    for (int cb = 0; cb < 8; ++cb)
      acc[cb] = __builtin_amdgcn_wmma_f32_16x16x32_bf16(
          false, a, false, b[cb], (short)0, acc[cb], false, false);
    a = a_next;
  }
  const int moff = (lane & 16) ? 8 : 0;
  for (int cb = 0; cb < 8; ++cb) {
    int col = cb * 16 + rit;
    float bia = (z == 3) ? bias[col] : 0.f;
    for (int v = 0; v < 8; ++v) {
      int r = r0 + moff + v;
      if (r < M) C[(size_t)r * FD + col] = acc[cb][v] + bia;
    }
  }
}

// ---------- edge scatter with per-(rel,dst) mean normalization ----------
// 32 threads per edge; each thread handles 4 consecutive feature dims (float4 message load).
__global__ void scatter_edges(const float* __restrict__ xw, const float* __restrict__ invdeg,
                              const int* __restrict__ src, const int* __restrict__ dst,
                              const int* __restrict__ et, float* __restrict__ agg, int E) {
  long long gid = (long long)blockIdx.x * blockDim.x + threadIdx.x;
  int e = (int)(gid >> 5);
  int q = (int)(gid & 31);
  if (e >= E) return;
  int s = src[e], t = dst[e], r = et[e];
  float nrm = invdeg[(size_t)r * NN + t];
  float4 m = *(const float4*)(xw + ((size_t)r * NN + s) * FD + 4 * q);
  float* ap = agg + (size_t)t * FD + 4 * q;
  atomicAdd(ap + 0, m.x * nrm);
  atomicAdd(ap + 1, m.y * nrm);
  atomicAdd(ap + 2, m.z * nrm);
  atomicAdd(ap + 3, m.w * nrm);
}

// ---------- f32 -> bf16 ----------
__global__ void conv_bf16(const float* __restrict__ src, unsigned short* __restrict__ dst, int n) {
  int i = blockIdx.x * blockDim.x + threadIdx.x;
  if (i < n) dst[i] = f2bf(src[i]);
}

// ---------- classifier head: out = h2[idx] @ fc3_w + fc3_b ----------
__global__ void final_head(const float* __restrict__ h2, const int* __restrict__ idx,
                           const float* __restrict__ w, const float* __restrict__ b,
                           float* __restrict__ out, int ni) {
  int t = blockIdx.x * blockDim.x + threadIdx.x;
  if (t >= ni * 2) return;
  int i = t >> 1, c = t & 1;
  const float* row = h2 + (size_t)idx[i] * FD;
  float s = b[c];
  for (int d = 0; d < FD; ++d) s += row[d] * w[d * 2 + c];
  out[t] = s;
}

extern "C" void kernel_launch(void* const* d_in, const int* in_sizes, int n_in,
                              void* d_out, int out_size, void* d_ws, size_t ws_size,
                              hipStream_t stream) {
  const float* value  = (const float*)d_in[0];
  const float* text   = (const float*)d_in[1];
  const float* fc1_w  = (const float*)d_in[2];
  const float* fc1_b  = (const float*)d_in[3];
  const float* fc2_w  = (const float*)d_in[4];
  const float* fc2_b  = (const float*)d_in[5];
  const float* relu_w = (const float*)d_in[6];
  const float* relu_b = (const float*)d_in[7];
  const float* w1rel  = (const float*)d_in[8];
  const float* w1root = (const float*)d_in[9];
  const float* b1     = (const float*)d_in[10];
  const float* w2rel  = (const float*)d_in[11];
  const float* w2root = (const float*)d_in[12];
  const float* b2     = (const float*)d_in[13];
  const float* fc3_w  = (const float*)d_in[14];
  const float* fc3_b  = (const float*)d_in[15];
  const int*   eidx   = (const int*)d_in[16];
  const int*   etype  = (const int*)d_in[17];
  const int*   idx    = (const int*)d_in[18];
  const int* esrc = eidx;
  const int* edst = eidx + NE;

  char* p = (char*)d_ws;
  float* bufA = (float*)p;              p += (size_t)NN * FD * 4;       // 25.6 MB
  float* bufB = (float*)p;              p += (size_t)NN * FD * 4;       // 25.6 MB
  unsigned short* hbf = (unsigned short*)p; p += (size_t)NN * FD * 2;   // 12.8 MB
  float* xw = (float*)p;                p += (size_t)NR * NN * FD * 4;  // 76.8 MB
  float* deg = (float*)p;               p += (size_t)NR * NN * 4;       // 0.6 MB
  float* Wc = (float*)p;                p += (size_t)KFUSE * FD * 4;
  float* biasc = (float*)p;             p += 512;
  unsigned* WcPack = (unsigned*)p;      p += (size_t)KFUSE * 64 * 4;
  unsigned* pack1 = (unsigned*)p;       p += (size_t)4 * FD * 64 * 4;
  unsigned* pack2 = (unsigned*)p;       p += (size_t)4 * FD * 64 * 4;

  // 1) fold fc1/fc2/relu_w into one encoder weight + pack all B matrices as WMMA fragments
  fuse_weights<<<(KFUSE * FD + FD + 255) / 256, 256, 0, stream>>>(
      fc1_w, fc1_b, fc2_w, fc2_b, relu_w, relu_b, Wc, biasc);
  pack_b<<<((KFUSE / 32) * 2048 + 255) / 256, 256, 0, stream>>>(Wc, WcPack, KFUSE);
  for (int r = 0; r < NR; ++r) {
    pack_b<<<32, 256, 0, stream>>>(w1rel + (size_t)r * FD * FD, pack1 + (size_t)r * 8192, FD);
    pack_b<<<32, 256, 0, stream>>>(w2rel + (size_t)r * FD * FD, pack2 + (size_t)r * 8192, FD);
  }
  pack_b<<<32, 256, 0, stream>>>(w1root, pack1 + (size_t)3 * 8192, FD);
  pack_b<<<32, 256, 0, stream>>>(w2root, pack2 + (size_t)3 * 8192, FD);

  // 2) per-(relation,dst) degree -> reciprocal (shared by both layers)
  zero_f32<<<(NR * NN + 255) / 256, 256, 0, stream>>>(deg, NR * NN);
  deg_count<<<(NE + 255) / 256, 256, 0, stream>>>(edst, etype, deg, NE);
  deg_inv<<<(NR * NN + 255) / 256, 256, 0, stream>>>(deg, NR * NN);

  // 3) fused encoder (bf16 WMMA, K=800): bufA = h (f32), hbf = h (bf16)
  gemm_fused_first<<<dim3((NN + 127) / 128), 256, 0, stream>>>(
      value, text, WcPack, biasc, bufA, hbf, NN);

  // 4) RGCN layer 1
  gemm_rgcn<<<dim3((NN + 127) / 128, 1, 4), 256, 0, stream>>>(hbf, pack1, b1, xw, bufB, NN);
  scatter_edges<<<(int)(((size_t)NE * 32 + 255) / 256), 256, 0, stream>>>(
      xw, deg, esrc, edst, etype, bufB, NE);
  conv_bf16<<<(NN * FD + 255) / 256, 256, 0, stream>>>(bufB, hbf, NN * FD);

  // 5) RGCN layer 2
  gemm_rgcn<<<dim3((NN + 127) / 128, 1, 4), 256, 0, stream>>>(hbf, pack2, b2, xw, bufA, NN);
  scatter_edges<<<(int)(((size_t)NE * 32 + 255) / 256), 256, 0, stream>>>(
      xw, deg, esrc, edst, etype, bufA, NE);

  // 6) head
  final_head<<<(NI * 2 + 255) / 256, 256, 0, stream>>>(bufA, idx, fc3_w, fc3_b, (float*)d_out, NI);
}
